// CustomFullyConnectedLayerGoogleTopK_65618510348678
// MI455X (gfx1250) — matrix operand order repaired and checked by reading, not compiled, not run on hardware
//
#include <hip/hip_runtime.h>

// ---------------------------------------------------------------------------
// Problem constants (reference: IN_F = OUT_F = 4096, batch 256, K = 3687)
// ---------------------------------------------------------------------------
#define BATCH 256
#define NF    4096          // OUT_F (GEMM N)
#define KF    4096          // IN_F  (GEMM K)
#define KCONST 3687.0f

#define NKB (KF / 32)       // 128 K-blocks of 32
#define NNB (NF / 16)       // 256 N-tiles of 16
#define NMB (BATCH / 16)    // 16  M-tiles of 16

typedef __attribute__((ext_vector_type(16))) __bf16 v16bf;
typedef __attribute__((ext_vector_type(8)))  float  v8f;

union Frag32B { v16bf v; uint4 q[2]; };             // one WMMA A/B operand per lane
union Pack16B { unsigned short h[8]; uint4 q; };    // 8 packed bf16 = 16B store

// round-to-nearest-even f32 -> bf16 (bit manipulation; no __bf16 scalar math)
__device__ __forceinline__ unsigned short f2bf(float f) {
    unsigned int u = __float_as_uint(f);
    unsigned int lsb = (u >> 16) & 1u;
    u += 0x7FFFu + lsb;
    return (unsigned short)(u >> 16);
}

// ---------------------------------------------------------------------------
// Pass 0: a[i] = clamp(K * softmax(alpha)[i], 0, 1)   (single block)
// ---------------------------------------------------------------------------
__global__ __launch_bounds__(256) void soft_topk_kernel(
    const float* __restrict__ alpha, float* __restrict__ a_out)
{
    __shared__ float red[256];
    const int tid = threadIdx.x;

    float m = -3.4e38f;
    for (int i = tid; i < NF; i += 256) m = fmaxf(m, alpha[i]);
    red[tid] = m; __syncthreads();
    for (int s = 128; s > 0; s >>= 1) {
        if (tid < s) red[tid] = fmaxf(red[tid], red[tid + s]);
        __syncthreads();
    }
    const float mx = red[0]; __syncthreads();

    float s = 0.0f;
    for (int i = tid; i < NF; i += 256) s += __expf(alpha[i] - mx);
    red[tid] = s; __syncthreads();
    for (int st = 128; st > 0; st >>= 1) {
        if (tid < st) red[tid] += red[tid + st];
        __syncthreads();
    }
    const float inv_sum = 1.0f / red[0];

    for (int i = tid; i < NF; i += 256) {
        float v = KCONST * __expf(alpha[i] - mx) * inv_sum;
        a_out[i] = fminf(1.0f, fmaxf(0.0f, v));
    }
}

// ---------------------------------------------------------------------------
// Pass 1a: repack x (f32, 256x4096) into WMMA A-fragment bf16 layout.
// A frag (mb, kb): per lane 16 bf16 contiguous (32B). Lane = (m&15) + 16*side,
// side/i per ISA 16-bit A 16x32 striping:
//   kk = k&31: side = (kk>>3)&1 ; i = (kk&7) + ((kk>>4)&1)*8
// hw offset = ((mb*NKB + kb)*32 + lane)*16 + i
// ---------------------------------------------------------------------------
__global__ __launch_bounds__(256) void pack_a_kernel(
    const float* __restrict__ x, unsigned short* __restrict__ Apack)
{
    const int g = blockIdx.x * 256 + threadIdx.x;   // 131072 threads total
    const int m = g >> 9;                           // 512 8-wide chunks per row
    const int k = (g & 511) * 8;

    const float4 f0 = *(const float4*)(x + (size_t)m * KF + k);
    const float4 f1 = *(const float4*)(x + (size_t)m * KF + k + 4);

    Pack16B p;
    p.h[0] = f2bf(f0.x); p.h[1] = f2bf(f0.y); p.h[2] = f2bf(f0.z); p.h[3] = f2bf(f0.w);
    p.h[4] = f2bf(f1.x); p.h[5] = f2bf(f1.y); p.h[6] = f2bf(f1.z); p.h[7] = f2bf(f1.w);

    const int mb   = m >> 4;
    const int kb   = k >> 5;
    const int kk   = k & 31;                 // multiple of 8
    const int side = (kk >> 3) & 1;
    const int i0   = ((kk >> 4) & 1) * 8;
    const int lane = (m & 15) + side * 16;
    const size_t off = ((size_t)(mb * NKB + kb) * 32 + lane) * 16 + i0;
    *(uint4*)(Apack + off) = p.q;
}

// ---------------------------------------------------------------------------
// Pass 1b: build the circulant weight W^T[c,r] = a[d]*V[d,c], d=(r-c)&4095,
// directly in WMMA B-fragment bf16 layout, staging a 95-row diagonal band of
// V through LDS (coalesced global reads, conflict-free stride-33 LDS).
// B frag (kb, nb): lane = (r&15) + 16*side, side = ((c&31)>>3)&1, i = (c&7) + (((c&31)>>4)&1)*8.
// hw offset = ((kb*NNB + nb)*32 + lane)*16 + i
// Block tile: 32 c-values (one kb) x 64 r-values (4 nb).
// ---------------------------------------------------------------------------
__global__ __launch_bounds__(256) void pack_b_kernel(
    const float* __restrict__ V, const float* __restrict__ a,
    unsigned short* __restrict__ Bpack)
{
    __shared__ float band[96 * 33];          // 95 needed rows (+1), pad stride 33
    const int tid = threadIdx.x;
    const int kb  = blockIdx.x;              // 0..127
    const int c0  = kb * 32;
    const int r0  = blockIdx.y * 64;         // 0..4032

    // Stage V band, pre-scaled by a[d]: d in [r0-c0-31, r0-c0+64]
    const int cl = tid & 31;
    for (int row = tid >> 5; row < 96; row += 8) {
        const int d = (r0 - c0 - 31 + row) & (KF - 1);
        band[row * 33 + cl] = V[(size_t)d * KF + c0 + cl] * a[d];
    }
    __syncthreads();

    // Each thread: 8 consecutive c for one r -> one 16B packed bf16 store.
    const int r_local = tid & 63;
    const int c_half  = tid >> 6;            // 0..3  (c_local = c_half*8 .. +7)
    Pack16B p;
#pragma unroll
    for (int j = 0; j < 8; ++j) {
        const int c_local = c_half * 8 + j;
        const int d_local = r_local - c_local + 31;        // 0..94
        p.h[j] = f2bf(band[d_local * 33 + c_local]);
    }
    const int r    = r0 + r_local;
    const int nb   = r >> 4;
    const int side = c_half & 1;
    const int lane = (r & 15) + side * 16;
    const int i0   = (c_half >> 1) * 8;
    const size_t off = ((size_t)(kb * NNB + nb) * 32 + lane) * 16 + i0;
    *(uint4*)(Bpack + off) = p.q;
}

// ---------------------------------------------------------------------------
// Pass 2: GEMM out(256x4096) = A(256x4096,bf16) * B(4096x4096,bf16)
// via v_wmma_f32_16x16x32_bf16. Operands already fragment-packed: every lane
// issues two coalesced global_load_b128 per fragment; reuse served by 192MB L2.
// Block = 256 threads = 8 waves (2 M-waves x 4 N-waves); wave tile 32M x 16N.
// Grid = (NF/64, BATCH/64) = (64, 4).
// ---------------------------------------------------------------------------
__global__ __launch_bounds__(256) void gemm_wmma_kernel(
    const unsigned short* __restrict__ Apack,
    const unsigned short* __restrict__ Bpack,
    float* __restrict__ out)
{
    const int tid    = threadIdx.x;
    const int lane   = tid & 31;
    const int wave   = tid >> 5;
    const int wave_m = wave >> 2;                    // 0..1
    const int wave_n = wave & 3;                     // 0..3
    const int mb0    = blockIdx.y * 4 + wave_m * 2;  // two M-tiles: mb0, mb0+1
    const int nb     = blockIdx.x * 4 + wave_n;      // one N-tile

    const unsigned short* aPtr0 = Apack + ((size_t)((mb0    ) * NKB) * 32 + lane) * 16;
    const unsigned short* aPtr1 = Apack + ((size_t)((mb0 + 1) * NKB) * 32 + lane) * 16;
    const unsigned short* bPtr  = Bpack + ((size_t)nb * 32 + lane) * 16;

    const size_t aStep = 32 * 16;                    // one kb in hw units
    const size_t bStep = (size_t)NNB * 32 * 16;

    v8f acc0 = {};
    v8f acc1 = {};

    for (int kb = 0; kb < NKB; ++kb) {
        Frag32B fa0, fa1, fb;
        const uint4* pa0 = (const uint4*)aPtr0;
        const uint4* pa1 = (const uint4*)aPtr1;
        const uint4* pb  = (const uint4*)bPtr;
        fa0.q[0] = pa0[0]; fa0.q[1] = pa0[1];
        fa1.q[0] = pa1[0]; fa1.q[1] = pa1[1];
        fb.q[0]  = pb[0];  fb.q[1]  = pb[1];

        // gfx1250 global_prefetch_b8 on the streaming B operand (2 kb ahead)
        __builtin_prefetch((const void*)(bPtr + 2 * bStep), 0, 1);

        acc0 = __builtin_amdgcn_wmma_f32_16x16x32_bf16(
                   false, fa0.v, false, fb.v, (short)0, acc0, false, false);
        acc1 = __builtin_amdgcn_wmma_f32_16x16x32_bf16(
                   false, fa1.v, false, fb.v, (short)0, acc1, false, false);

        aPtr0 += aStep; aPtr1 += aStep; bPtr += bStep;
    }

    // C/D layout (ISA 7.12.2): VGPR v -> M = v + 8*(lane>>4), N = lane&15
    const int col = nb * 16 + (lane & 15);
    const int rhi = (lane >> 4) * 8;
    float* o0 = out + (size_t)(mb0 * 16 + rhi) * NF + col;
    float* o1 = out + (size_t)((mb0 + 1) * 16 + rhi) * NF + col;
#pragma unroll
    for (int v = 0; v < 8; ++v) {
        o0[(size_t)v * NF] = acc0[v];
        o1[(size_t)v * NF] = acc1[v];
    }
}

// ---------------------------------------------------------------------------
// Host-side launcher
// ---------------------------------------------------------------------------
extern "C" void kernel_launch(void* const* d_in, const int* in_sizes, int n_in,
                              void* d_out, int out_size, void* d_ws, size_t ws_size,
                              hipStream_t stream)
{
    (void)in_sizes; (void)n_in; (void)out_size; (void)ws_size;

    const float* x     = (const float*)d_in[0];   // (256, 4096)
    const float* V     = (const float*)d_in[1];   // (4096, 4096)
    const float* alpha = (const float*)d_in[2];   // (4096,)
    float*       out   = (float*)d_out;           // (256, 4096)

    char* ws = (char*)d_ws;
    unsigned short* Bpack = (unsigned short*)ws;                              // 32 MiB
    unsigned short* Apack = (unsigned short*)(ws + (size_t)KF * NF * 2);      //  2 MiB
    float*          a_ws  = (float*)(ws + (size_t)KF * NF * 2
                                        + (size_t)BATCH * KF * 2);            // 16 KiB

    soft_topk_kernel<<<1, 256, 0, stream>>>(alpha, a_ws);
    pack_a_kernel<<<dim3((BATCH * KF / 8) / 256), 256, 0, stream>>>(x, Apack);
    pack_b_kernel<<<dim3(NKB, NF / 64), 256, 0, stream>>>(V, a_ws, Bpack);
    gemm_wmma_kernel<<<dim3(NF / 64, BATCH / 64), 256, 0, stream>>>(Apack, Bpack, out);
}
